// RnnTimeSeriesEncoder_24550033064431
// MI455X (gfx1250) — compile-verified, matrix operand order checked by reading, not run on hardware
//
#include <hip/hip_runtime.h>

typedef __attribute__((ext_vector_type(16))) __bf16 v16bf;
typedef __attribute__((ext_vector_type(8)))  float  v8f;

#define HID   256
#define TLEN  512
#define IN0   32

__device__ __forceinline__ unsigned short f2bf(float f) {
  union { float f; unsigned u; } x; x.f = f;
  return (unsigned short)(x.u >> 16);
}

union Frag {
  v16bf v;
  unsigned short s[16];
  uint4 q[2];
};

// 32 contiguous bytes -> one 16xbf16 fragment (B operand / packed-x A operand)
__device__ __forceinline__ v16bf load16(const unsigned short* p) {
  Frag r;
  r.q[0] = *(const uint4*)(p);
  r.q[1] = *(const uint4*)(p + 8);
  return r.v;
}

// A operand from LDS row-major h[16][256] bf16.
// lane<16: row=lane,    k in {kk*32+0..7, kk*32+16..23}
// lane>=16: row=lane-16, k shifted by +8  (mirrors 16-bit A-matrix VGPR layout)
__device__ __forceinline__ v16bf load_a_lds(const unsigned short* p) {
  Frag r;
  r.q[0] = *(const uint4*)(p);       // k .. k+7
  r.q[1] = *(const uint4*)(p + 16);  // k+16 .. k+23
  return r.v;
}

__device__ __forceinline__ float sigm(float x) { return 1.0f / (1.0f + __expf(-x)); }
__device__ __forceinline__ float tanhfast(float x) {
  float e = __expf(2.0f * x);
  return (e - 1.0f) / (e + 1.0f);
}

// ---------------------------------------------------------------------------
// Pack W[4H x K] fp32 (row = output n, col = k) into bf16 WMMA B-operand tiles:
// dst[tile(nt*nkt+kt)][lane][j], lane: n_local = lane&15, hi = lane>>4,
// j -> k_local = j + (j>=8 ? 8 : 0) + hi*8   (two 8-half runs, +8 for hi lanes)
// ---------------------------------------------------------------------------
__global__ void pack_w_kernel(const float* __restrict__ W, unsigned short* __restrict__ dst,
                              int K, int nkt) {
  int e = blockIdx.x * 256 + threadIdx.x;
  int total = 64 * nkt * 512;
  if (e >= total) return;
  int tile = e >> 9;
  int r    = e & 511;
  int lane = r >> 4;
  int j    = r & 15;
  int nt = tile / nkt;
  int kt = tile % nkt;
  int nl = lane & 15;
  int hi = lane >> 4;
  int kl = j + ((j >> 3) << 3) + hi * 8;
  int n = nt * 16 + nl;
  int k = kt * 32 + kl;
  dst[tile * 512 + lane * 16 + j] = f2bf(W[n * K + k]);
}

// ---------------------------------------------------------------------------
// Pack x: M[B,T,32] fp32 -> bf16 A-fragment layout, one 512-half record per
// (batch-tile, t):  dst[((wg*T + t)*32 + lane)*16 + j]
// ---------------------------------------------------------------------------
__global__ void pack_x_kernel(const float* __restrict__ M, unsigned short* __restrict__ dst) {
  int e = blockIdx.x * 256 + threadIdx.x;    // over 16*512*32*16 = 4194304
  int j    = e & 15;
  int lane = (e >> 4) & 31;
  int t    = (e >> 9) & (TLEN - 1);
  int wg   = e >> 18;
  int nl = lane & 15;
  int hi = lane >> 4;
  int k  = j + ((j >> 3) << 3) + hi * 8;     // 0..31
  int b  = wg * 16 + nl;
  dst[e] = f2bf(M[((size_t)b * TLEN + t) * IN0 + k]);
}

// ---------------------------------------------------------------------------
// Persistent fused 2-layer LSTM. grid = B/16 workgroups, 512 threads (16 waves).
// Wave w owns hidden columns [16w, 16w+16) for all four gates
// (N-tiles g*16+w, g = i,f,g,o), so the cell update is wave-local.
// ---------------------------------------------------------------------------
__global__ __launch_bounds__(512) void lstm2_fused_kernel(
    const unsigned short* __restrict__ pX,
    const unsigned short* __restrict__ pWih0,
    const unsigned short* __restrict__ pWhh0,
    const unsigned short* __restrict__ pWih1,
    const unsigned short* __restrict__ pWhh1,
    const float* __restrict__ bih0, const float* __restrict__ bhh0,
    const float* __restrict__ bih1, const float* __restrict__ bhh1,
    float* __restrict__ out) {
  __shared__ unsigned short h0s[16 * HID];   // 8 KB, bf16 h of layer 0
  __shared__ unsigned short h1s[16 * HID];   // 8 KB, bf16 h of layer 1
  __shared__ float bias0[4 * HID];           // 4 KB combined biases
  __shared__ float bias1[4 * HID];

  const int tid   = threadIdx.x;
  const int w     = tid >> 5;       // wave 0..15
  const int lane  = tid & 31;
  const int nl    = lane & 15;
  const int hi    = lane >> 4;
  const int col   = w * 16 + nl;            // hidden column this lane owns
  const int bbase = blockIdx.x * 16;        // batch rows [bbase, bbase+16)

  for (int i = tid; i < 4 * HID; i += 512) {
    bias0[i] = bih0[i] + bhh0[i];
    bias1[i] = bih1[i] + bhh1[i];
  }
  for (int i = tid; i < 16 * HID; i += 512) { h0s[i] = 0; h1s[i] = 0; }
  __syncthreads();

  v8f c0 = {};  // layer-0 cell state, rows r+hi*8, column `col`
  v8f c1 = {};  // layer-1 cell state

  const unsigned short* vX = pX + (size_t)blockIdx.x * TLEN * 512;

  // Opaque zero offset, redefined every iteration: defeats LICM of the
  // loop-invariant weight-fragment loads (which would spill to scratch)
  // while keeping the bases as true global pointers -> global_load_b128
  // with SGPR base + VGPR offset, decoupled LOADcnt/DScnt.
  int zoff = 0;

  for (int t = 0; t < TLEN; ++t) {
    asm volatile("" : "+s"(zoff));
    const unsigned short* wX    = vX + zoff;
    const unsigned short* wih0p = pWih0 + zoff;
    const unsigned short* whh0p = pWhh0 + zoff;
    const unsigned short* wih1p = pWih1 + zoff;
    const unsigned short* whh1p = pWhh1 + zoff;

    // ================= layer 0: gates = bias + x_t@Wih0^T + h0@Whh0^T =======
    v8f acc[4];
#pragma unroll
    for (int g = 0; g < 4; ++g) {
      float bv = bias0[g * HID + col];
#pragma unroll
      for (int r = 0; r < 8; ++r) acc[g][r] = bv;
    }
    {
      // x_t: pre-packed bf16 fragment, single K-step
      v16bf xa = load16(wX + (size_t)t * 512 + lane * 16);
#pragma unroll
      for (int g = 0; g < 4; ++g) {
        v16bf b = load16(wih0p + (size_t)(g * 16 + w) * 512 + lane * 16);
        acc[g] = __builtin_amdgcn_wmma_f32_16x16x32_bf16(false, xa, false, b,
                                                         (short)0, acc[g], false, false);
      }
    }
    {
      // h0_{t-1} contribution, ping-pong pipelined B-fragment loads
      v16bf bb[2][4];
#pragma unroll
      for (int g = 0; g < 4; ++g)
        bb[0][g] = load16(whh0p + (size_t)((g * 16 + w) * 8 + 0) * 512 + lane * 16);
#pragma unroll
      for (int kk = 0; kk < 8; ++kk) {
        const int cur = kk & 1, nxt = cur ^ 1;
        if (kk < 7) {
#pragma unroll
          for (int g = 0; g < 4; ++g)
            bb[nxt][g] = load16(whh0p + (size_t)((g * 16 + w) * 8 + kk + 1) * 512 + lane * 16);
        }
        v16bf a = load_a_lds(&h0s[nl * HID + kk * 32 + hi * 8]);
#pragma unroll
        for (int g = 0; g < 4; ++g)
          acc[g] = __builtin_amdgcn_wmma_f32_16x16x32_bf16(false, a, false, bb[cur][g],
                                                           (short)0, acc[g], false, false);
      }
    }
    __syncthreads();  // everyone done reading h0_{t-1}

    // pointwise layer 0: c0 = f*c0 + i*g ; h0 = o*tanh(c0) -> LDS (bf16)
#pragma unroll
    for (int r = 0; r < 8; ++r) {
      float iv = sigm(acc[0][r]);
      float fv = sigm(acc[1][r]);
      float gv = tanhfast(acc[2][r]);
      float ov = sigm(acc[3][r]);
      float c  = fv * c0[r] + iv * gv;
      c0[r] = c;
      float h = ov * tanhfast(c);
      h0s[(r + hi * 8) * HID + col] = f2bf(h);
    }
    __syncthreads();  // h0_t visible

    // ================= layer 1: gates = bias + h0_t@Wih1^T + h1@Whh1^T ======
#pragma unroll
    for (int g = 0; g < 4; ++g) {
      float bv = bias1[g * HID + col];
#pragma unroll
      for (int r = 0; r < 8; ++r) acc[g][r] = bv;
    }
    {
      v16bf bb[2][4];
#pragma unroll
      for (int g = 0; g < 4; ++g)
        bb[0][g] = load16(wih1p + (size_t)((g * 16 + w) * 8 + 0) * 512 + lane * 16);
#pragma unroll
      for (int kk = 0; kk < 8; ++kk) {
        const int cur = kk & 1, nxt = cur ^ 1;
        if (kk < 7) {
#pragma unroll
          for (int g = 0; g < 4; ++g)
            bb[nxt][g] = load16(wih1p + (size_t)((g * 16 + w) * 8 + kk + 1) * 512 + lane * 16);
        }
        v16bf a = load_a_lds(&h0s[nl * HID + kk * 32 + hi * 8]);
#pragma unroll
        for (int g = 0; g < 4; ++g)
          acc[g] = __builtin_amdgcn_wmma_f32_16x16x32_bf16(false, a, false, bb[cur][g],
                                                           (short)0, acc[g], false, false);
      }
    }
    {
      v16bf bb[2][4];
#pragma unroll
      for (int g = 0; g < 4; ++g)
        bb[0][g] = load16(whh1p + (size_t)((g * 16 + w) * 8 + 0) * 512 + lane * 16);
#pragma unroll
      for (int kk = 0; kk < 8; ++kk) {
        const int cur = kk & 1, nxt = cur ^ 1;
        if (kk < 7) {
#pragma unroll
          for (int g = 0; g < 4; ++g)
            bb[nxt][g] = load16(whh1p + (size_t)((g * 16 + w) * 8 + kk + 1) * 512 + lane * 16);
        }
        v16bf a = load_a_lds(&h1s[nl * HID + kk * 32 + hi * 8]);
#pragma unroll
        for (int g = 0; g < 4; ++g)
          acc[g] = __builtin_amdgcn_wmma_f32_16x16x32_bf16(false, a, false, bb[cur][g],
                                                           (short)0, acc[g], false, false);
      }
    }
    __syncthreads();  // everyone done reading h0_t / h1_{t-1}

    // pointwise layer 1
#pragma unroll
    for (int r = 0; r < 8; ++r) {
      float iv = sigm(acc[0][r]);
      float fv = sigm(acc[1][r]);
      float gv = tanhfast(acc[2][r]);
      float ov = sigm(acc[3][r]);
      float c  = fv * c1[r] + iv * gv;
      c1[r] = c;
      float h = ov * tanhfast(c);
      h1s[(r + hi * 8) * HID + col] = f2bf(h);
      if (t == TLEN - 1) {
        out[(size_t)(bbase + r + hi * 8) * HID + col] = h;
      }
    }
    __syncthreads();  // h1_t visible
  }
}

extern "C" void kernel_launch(void* const* d_in, const int* in_sizes, int n_in,
                              void* d_out, int out_size, void* d_ws, size_t ws_size,
                              hipStream_t stream) {
  const float* M     = (const float*)d_in[0];
  const float* W_ih0 = (const float*)d_in[1];
  const float* W_hh0 = (const float*)d_in[2];
  const float* b_ih0 = (const float*)d_in[3];
  const float* b_hh0 = (const float*)d_in[4];
  const float* W_ih1 = (const float*)d_in[5];
  const float* W_hh1 = (const float*)d_in[6];
  const float* b_ih1 = (const float*)d_in[7];
  const float* b_hh1 = (const float*)d_in[8];

  unsigned short* ws = (unsigned short*)d_ws;
  unsigned short* p_wih0 = ws;                          // 64*1*512 halves
  unsigned short* p_whh0 = p_wih0 + 64 * 1 * 512;       // 64*8*512
  unsigned short* p_wih1 = p_whh0 + 64 * 8 * 512;       // 64*8*512
  unsigned short* p_whh1 = p_wih1 + 64 * 8 * 512;       // 64*8*512
  unsigned short* p_x    = p_whh1 + 64 * 8 * 512;       // 16*512*512 halves (8 MB)

  pack_w_kernel<<<(64 * 1 * 512 + 255) / 256, 256, 0, stream>>>(W_ih0, p_wih0, IN0, 1);
  pack_w_kernel<<<(64 * 8 * 512 + 255) / 256, 256, 0, stream>>>(W_hh0, p_whh0, HID, 8);
  pack_w_kernel<<<(64 * 8 * 512 + 255) / 256, 256, 0, stream>>>(W_ih1, p_wih1, HID, 8);
  pack_w_kernel<<<(64 * 8 * 512 + 255) / 256, 256, 0, stream>>>(W_hh1, p_whh1, HID, 8);
  pack_x_kernel<<<(16 * TLEN * 512) / 256, 256, 0, stream>>>(M, p_x);

  lstm2_fused_kernel<<<256 / 16, 512, 0, stream>>>(
      p_x, p_wih0, p_whh0, p_wih1, p_whh1,
      b_ih0, b_hh0, b_ih1, b_hh1, (float*)d_out);
}
